// SpatialAttention_28432683499776
// MI455X (gfx1250) — compile-verified, matrix operand order checked by reading, not run on hardware
//
#include <hip/hip_runtime.h>
#include <hip/hip_bf16.h>

// ---------------------------------------------------------------------------
// SpatialAttention (GroupNorm -> QKV -> softmax(QK^T/sqrt(C)) V -> proj + res)
// B=4, C=512, H=W=64 (AREA=4096), G=32 groups.  All contractions run on
// v_wmma_f32_16x16x32_bf16 (wave32 CDNA5 matrix path); attention is computed
// flash-style (no 4k x 4k score matrix materialization).
//
// v2: double-buffered + software-pipelined LDS staging in the dense GEMMs,
//     parallelized online-softmax (16 rows x 8 segments), global_prefetch of
//     the next K/V tile during the WMMA phase of flash attention.
// ---------------------------------------------------------------------------

constexpr int Bn   = 4;
constexpr int Cc   = 512;
constexpr int Gg   = 32;
constexpr int CG   = Cc / Gg;     // 16 channels per group
constexpr int AREA = 64 * 64;     // 4096
constexpr int BG   = Bn * Gg;     // 128 (b,g) pairs

typedef __bf16 bf16;
typedef __attribute__((ext_vector_type(16))) __bf16 v16bf;
typedef __attribute__((ext_vector_type(8)))  float  v8f;

__device__ inline v8f v8f_zero() {
    v8f z = {0.f, 0.f, 0.f, 0.f, 0.f, 0.f, 0.f, 0.f};
    return z;
}

__device__ inline v8f wmma_bf16(v16bf a, v16bf b, v8f c) {
    // D = A(16x32) * B(32x16) + C, f32 accumulate
    return __builtin_amdgcn_wmma_f32_16x16x32_bf16(
        /*neg_a=*/false, a, /*neg_b=*/false, b,
        /*c_mod=*/(short)0, c, /*reuse_a=*/false, /*reuse_b=*/false);
}

// ---------------------------------------------------------------------------
// Kernel 0: fp32 -> bf16 convert of the four 512x512 weight matrices
// wbf layout: [wq | wk | wv | wo], each row-major [o][c] (K=c contiguous)
// ---------------------------------------------------------------------------
__global__ __launch_bounds__(256)
void convert_weights(const float* wq, const float* wk, const float* wv,
                     const float* wo, bf16* wbf) {
    int m = blockIdx.y;
    const float* src = (m == 0) ? wq : (m == 1) ? wk : (m == 2) ? wv : wo;
    int idx = blockIdx.x * 256 + threadIdx.x;           // 512*512 / 256 = 1024 blocks
    wbf[(size_t)m * Cc * Cc + idx] = (bf16)src[idx];
}

// ---------------------------------------------------------------------------
// Kernel 1: GroupNorm statistics.  One block per (b,g); channels of a group
// are contiguous, so the group is one contiguous 16*4096 fp32 span.
// stats[0..BG)   = mean,  stats[BG..2BG) = rstd
// ---------------------------------------------------------------------------
__global__ __launch_bounds__(256)
void gn_stats(const float* net, float* stats) {
    int bg = blockIdx.x;
    const float* base = net + (size_t)bg * (CG * AREA);
    __shared__ float ssum[256], ssq[256];
    float s = 0.f, q = 0.f;
    for (int i = threadIdx.x; i < CG * AREA; i += 256) {
        float x = base[i];
        s += x; q += x * x;
    }
    ssum[threadIdx.x] = s; ssq[threadIdx.x] = q;
    __syncthreads();
    for (int st = 128; st > 0; st >>= 1) {
        if (threadIdx.x < st) {
            ssum[threadIdx.x] += ssum[threadIdx.x + st];
            ssq[threadIdx.x]  += ssq[threadIdx.x + st];
        }
        __syncthreads();
    }
    if (threadIdx.x == 0) {
        const float inv_n = 1.f / (float)(CG * AREA);
        float mean = ssum[0] * inv_n;
        float var  = ssq[0] * inv_n - mean * mean;
        stats[bg]      = mean;
        stats[BG + bg] = __frsqrt_rn(var + 1e-6f);
    }
}

// ---------------------------------------------------------------------------
// Kernel 2: apply GroupNorm affine and transpose to xnT[b][a][c] (bf16,
// c-contiguous) via a 64x64 LDS tile so both global accesses are coalesced.
// ---------------------------------------------------------------------------
__global__ __launch_bounds__(256)
void gn_apply_transpose(const float* net, const float* gscale, const float* gbias,
                        const float* stats, bf16* xnT) {
    __shared__ bf16 tile[64 * 72];          // pad 64 -> 72 to spread banks
    int b  = blockIdx.z;
    int c0 = blockIdx.y * 64;
    int a0 = blockIdx.x * 64;
    const float* src = net + ((size_t)b * Cc + c0) * AREA + a0;
    for (int i = 0; i < 16; ++i) {
        int idx = threadIdx.x + i * 256;
        int cc = idx >> 6, aa = idx & 63;
        int c = c0 + cc;
        int g = c >> 4;                      // CG = 16
        float mu = stats[b * Gg + g];
        float rs = stats[BG + b * Gg + g];
        float x  = src[(size_t)cc * AREA + aa];
        tile[cc * 72 + aa] = (bf16)(((x - mu) * rs) * gscale[c] + gbias[c]);
    }
    __syncthreads();
    bf16* dst = xnT + ((size_t)b * AREA + a0) * Cc + c0;
    for (int i = 0; i < 16; ++i) {
        int idx = threadIdx.x + i * 256;
        int aa = idx >> 6, cc = idx & 63;
        dst[(size_t)aa * Cc + cc] = tile[cc * 72 + aa];
    }
}

// ---------------------------------------------------------------------------
// Shared 128x128 NT-GEMM core (K = 512, bf16 in, f32 accum), double-buffered.
// A: [M][512] row-major, B: [N][512] row-major (i.e. D = A * B^T).
// 256 threads = 8 waves; wave w owns output tile-row w (16 rows) x 8 col-tiles.
// Software pipeline: chunk kc+1 global loads are issued before the chunk-kc
// WMMA block, drained into the alternate LDS buffer after it; 1 barrier/chunk.
// ---------------------------------------------------------------------------
__device__ inline void gemm128_db(const bf16* __restrict__ Aop,
                                  const bf16* __restrict__ Bop,
                                  int m0, int n0,
                                  bf16* As0, bf16* Bs0, bf16* As1, bf16* Bs1,
                                  v8f (&acc)[8]) {
    const int tid  = threadIdx.x;
    const int lane = tid & 31;
    const int w    = tid >> 5;
    const int lr   = lane & 15;
    const int lh   = (lane >> 4) << 4;       // 0 or 16 (K half select)

    for (int nt = 0; nt < 8; ++nt) acc[nt] = v8f_zero();

    // per-thread staging coordinates: 4 uint4 (8 bf16 each) per matrix per chunk
    int srow[4], scol[4];
    #pragma unroll
    for (int i = 0; i < 4; ++i) {
        int idx = tid + i * 256;
        srow[i] = idx >> 3;
        scol[i] = (idx & 7) * 8;
    }

    uint4 ra[4], rb[4];
    // preload chunk 0
    #pragma unroll
    for (int i = 0; i < 4; ++i) {
        ra[i] = *(const uint4*)&Aop[(size_t)(m0 + srow[i]) * Cc + scol[i]];
        rb[i] = *(const uint4*)&Bop[(size_t)(n0 + srow[i]) * Cc + scol[i]];
    }
    #pragma unroll
    for (int i = 0; i < 4; ++i) {
        *(uint4*)&As0[srow[i] * 72 + scol[i]] = ra[i];
        *(uint4*)&Bs0[srow[i] * 72 + scol[i]] = rb[i];
    }
    __syncthreads();

    for (int kc = 0; kc < 8; ++kc) {         // 512 / 64
        bf16* Ac = (kc & 1) ? As1 : As0;
        bf16* Bc = (kc & 1) ? Bs1 : Bs0;
        bf16* An = (kc & 1) ? As0 : As1;
        bf16* Bnx = (kc & 1) ? Bs0 : Bs1;

        if (kc < 7) {                        // issue next chunk's global loads
            int k1 = (kc + 1) * 64;
            #pragma unroll
            for (int i = 0; i < 4; ++i) {
                ra[i] = *(const uint4*)&Aop[(size_t)(m0 + srow[i]) * Cc + k1 + scol[i]];
                rb[i] = *(const uint4*)&Bop[(size_t)(n0 + srow[i]) * Cc + k1 + scol[i]];
            }
        }

        #pragma unroll
        for (int s = 0; s < 2; ++s) {        // two K=32 steps
            v16bf af = *(const v16bf*)&Ac[(w * 16 + lr) * 72 + s * 32 + lh];
            #pragma unroll
            for (int nt = 0; nt < 8; ++nt) {
                v16bf bfr = *(const v16bf*)&Bc[(nt * 16 + lr) * 72 + s * 32 + lh];
                acc[nt] = wmma_bf16(af, bfr, acc[nt]);
            }
        }

        if (kc < 7) {                        // drain into alternate buffer
            #pragma unroll
            for (int i = 0; i < 4; ++i) {
                *(uint4*)&An[srow[i] * 72 + scol[i]]  = ra[i];
                *(uint4*)&Bnx[srow[i] * 72 + scol[i]] = rb[i];
            }
        }
        __syncthreads();
    }
}

constexpr size_t GEMM_DB_BYTES = (size_t)4 * (128 * 72) * 2;     // 73,728 B

// ---------------------------------------------------------------------------
// Kernel 3: QKV projections.
//  mode 0/1: qT/kT[b][a][o] = xnT[b][a][:] . w[o][:] + bias[o]   (D tile [a][o])
//  mode 2:   v[b][o][a]     = wv[o][:] . xnT[b][a][:] + bias[o]  (D tile [o][a])
// ---------------------------------------------------------------------------
__global__ __launch_bounds__(256)
void qkv_gemm(const bf16* __restrict__ xnT, const bf16* __restrict__ wbf,
              const float* bq, const float* bk, const float* bv,
              bf16* qT, bf16* kT, bf16* vM) {
    extern __shared__ char smem[];
    bf16* As0 = (bf16*)smem;
    bf16* Bs0 = As0 + 128 * 72;
    bf16* As1 = Bs0 + 128 * 72;
    bf16* Bs1 = As1 + 128 * 72;

    int z = blockIdx.z;
    int b = z / 3, mode = z % 3;
    const bf16* xb = xnT + (size_t)b * AREA * Cc;
    const bf16* wm = wbf + (size_t)mode * Cc * Cc;   // wq,wk,wv in order

    const bf16 *Aop, *Bop;
    const float* bias;
    bf16* Dout;
    int m0, n0;
    size_t ldD;
    bool bias_by_row;
    if (mode < 2) {
        Aop = xb;  m0 = blockIdx.x * 128;            // M = a (32 tiles)
        Bop = wm;  n0 = blockIdx.y * 128;            // N = o (4 tiles)
        Dout = ((mode == 0) ? qT : kT) + (size_t)b * AREA * Cc;
        ldD = Cc; bias = (mode == 0) ? bq : bk; bias_by_row = false;
    } else {
        Aop = wm;  m0 = blockIdx.y * 128;            // M = o
        Bop = xb;  n0 = blockIdx.x * 128;            // N = a
        Dout = vM + (size_t)b * Cc * AREA;
        ldD = AREA; bias = bv; bias_by_row = true;
    }

    v8f acc[8];
    gemm128_db(Aop, Bop, m0, n0, As0, Bs0, As1, Bs1, acc);

    const int lane = threadIdx.x & 31;
    const int w    = threadIdx.x >> 5;
    const int lr   = lane & 15;
    #pragma unroll
    for (int nt = 0; nt < 8; ++nt) {
        #pragma unroll
        for (int r = 0; r < 8; ++r) {
            int m = w * 16 + r + 8 * (lane >> 4);
            int n = nt * 16 + lr;
            float bval = bias_by_row ? bias[m0 + m] : bias[n0 + n];
            Dout[(size_t)(m0 + m) * ldD + (n0 + n)] = (bf16)(acc[nt][r] + bval);
        }
    }
}

// ---------------------------------------------------------------------------
// Kernel 4: flash attention.  One block per (query-tile of 16, batch).
// Iterates keys in 128-wide tiles: S = Q K^T (WMMA), online softmax
// (parallel 16x8 segmented reduction), O += P V (WMMA).  Output oT[b][a][c].
// Dynamic LDS ~295 KB (WGP budget 320 KB).  The next K/V tile is pulled
// toward the WGP with global_prefetch during the WMMA phase.
// ---------------------------------------------------------------------------
constexpr int QS_LD = 520;   // 512 + 8 bf16 pad (keeps rows 16B aligned)
constexpr int KS_LD = 520;
constexpr int VS_LD = 136;   // 128 + 8
constexpr int PS_LD = 136;
constexpr int SS_LD = 132;   // f32 row pad

constexpr size_t ATTN_SMEM_BYTES =
    (size_t)(16 * QS_LD + 128 * KS_LD + 512 * VS_LD + 16 * PS_LD) * 2  // bf16
    + (size_t)(16 * SS_LD) * 4                                         // Ss f32
    + 320 * 4;                                                         // stats+red

__global__ __launch_bounds__(256)
void flash_attn(const bf16* __restrict__ qT, const bf16* __restrict__ kT,
                const bf16* __restrict__ vM, bf16* __restrict__ oT) {
    extern __shared__ char smem[];
    bf16*  Qs   = (bf16*)smem;
    bf16*  Ks   = Qs + 16 * QS_LD;
    bf16*  Vs   = Ks + 128 * KS_LD;
    bf16*  Ps   = Vs + 512 * VS_LD;
    float* Ss   = (float*)(Ps + 16 * PS_LD);
    float* m_sh = Ss + 16 * SS_LD;       // [16] running row max
    float* s_sh = m_sh + 16;             // [16] running row sum
    float* cf   = s_sh + 16;             // [16] correction factor this tile
    float* redm = cf + 16;               // [16][8] segment max scratch
    float* reds = redm + 128;            // [16][8] segment sum scratch

    const int b   = blockIdx.y;
    const int i0  = blockIdx.x * 16;
    const int tid = threadIdx.x;
    const int lane = tid & 31;
    const int w    = tid >> 5;
    const int lr   = lane & 15;
    const int lh   = (lane >> 4) << 4;

    const bf16* qb = qT + ((size_t)b * AREA + i0) * Cc;
    const bf16* kb = kT + (size_t)b * AREA * Cc;
    const bf16* vb = vM + (size_t)b * Cc * AREA;

    // stage the 16x512 query tile once
    for (int i = 0; i < 4; ++i) {
        int idx = tid + i * 256;
        int row = idx >> 6;
        int c8  = (idx & 63) * 8;
        *(uint4*)&Qs[row * QS_LD + c8] = *(const uint4*)&qb[(size_t)row * Cc + c8];
    }
    if (tid < 16) { m_sh[tid] = -3.0e38f; s_sh[tid] = 0.f; }

    v8f Oacc[4];
    #pragma unroll
    for (int ct = 0; ct < 4; ++ct) Oacc[ct] = v8f_zero();

    const float scale = 0.044194173824159216f;   // 1/sqrt(512)

    for (int j0 = 0; j0 < AREA; j0 += 128) {
        __syncthreads();                         // protect Ks/Vs/Ps from prev iter
        // stage K tile: 128 rows x 512 bf16  (kT is [a][c], c-contiguous)
        for (int i = 0; i < 32; ++i) {
            int idx = tid + i * 256;
            int row = idx >> 6;
            int c8  = (idx & 63) * 8;
            *(uint4*)&Ks[row * KS_LD + c8] =
                *(const uint4*)&kb[((size_t)(j0 + row)) * Cc + c8];
        }
        // stage V tile: 512 rows x 128 bf16  (v is [c][a], a-contiguous)
        for (int i = 0; i < 32; ++i) {
            int idx = tid + i * 256;
            int row = idx >> 4;
            int c8  = (idx & 15) * 8;
            *(uint4*)&Vs[row * VS_LD + c8] =
                *(const uint4*)&vb[(size_t)row * AREA + j0 + c8];
        }
        __syncthreads();

        // prefetch next j-tile (K: 1024 cachelines, V: 1024 half-rows) while
        // the matrix cores chew on this one
        if (j0 + 128 < AREA) {
            const char* kn = (const char*)&kb[(size_t)(j0 + 128) * Cc];
            #pragma unroll
            for (int i = 0; i < 4; ++i) {
                int idx = tid + i * 256;
                __builtin_prefetch(kn + (size_t)idx * 128, 0, 1);
            }
            #pragma unroll
            for (int i = 0; i < 4; ++i) {
                int idx = tid + i * 256;
                int row = idx >> 1;
                __builtin_prefetch((const char*)&vb[(size_t)row * AREA + j0 + 128]
                                       + (idx & 1) * 128, 0, 1);
            }
        }

        // S tile: wave w computes the 16x16 score block for keys j0+16w..+15
        v8f sacc = v8f_zero();
        #pragma unroll
        for (int ks = 0; ks < 16; ++ks) {        // K = 512 in 32-steps
            v16bf aq  = *(const v16bf*)&Qs[lr * QS_LD + ks * 32 + lh];
            v16bf bk2 = *(const v16bf*)&Ks[(w * 16 + lr) * KS_LD + ks * 32 + lh];
            sacc = wmma_bf16(aq, bk2, sacc);
        }
        #pragma unroll
        for (int r = 0; r < 8; ++r) {
            int m = r + 8 * (lane >> 4);
            int n = w * 16 + lr;
            Ss[m * SS_LD + n] = sacc[r] * scale;
        }
        __syncthreads();

        // ---- parallel online softmax: 16 rows x 8 segments of 16 cols ----
        if (tid < 128) {                         // segment max
            int r = tid >> 3, sg = tid & 7;
            float mx = -3.0e38f;
            #pragma unroll
            for (int jj = 0; jj < 16; ++jj)
                mx = fmaxf(mx, Ss[r * SS_LD + sg * 16 + jj]);
            redm[tid] = mx;
        }
        __syncthreads();
        if (tid < 16) {                          // new row max + correction
            float mold = m_sh[tid];
            float mx = mold;
            #pragma unroll
            for (int sg = 0; sg < 8; ++sg) mx = fmaxf(mx, redm[tid * 8 + sg]);
            m_sh[tid] = mx;
            cf[tid] = __expf(mold - mx);         // 0 on first tile (underflow)
        }
        __syncthreads();
        if (tid < 128) {                         // exp + segment sum, P in bf16
            int r = tid >> 3, sg = tid & 7;
            float mx = m_sh[r];
            float sum = 0.f;
            #pragma unroll
            for (int jj = 0; jj < 16; ++jj) {
                float p = __expf(Ss[r * SS_LD + sg * 16 + jj] - mx);
                Ps[r * PS_LD + sg * 16 + jj] = (bf16)p;
                sum += p;
            }
            reds[tid] = sum;
        }
        __syncthreads();
        if (tid < 16) {                          // running denominator
            float sum = 0.f;
            #pragma unroll
            for (int sg = 0; sg < 8; ++sg) sum += reds[tid * 8 + sg];
            s_sh[tid] = s_sh[tid] * cf[tid] + sum;
        }
        __syncthreads();

        // rescale running O accumulators, then O += P * V (wave w: c-tiles 4w..4w+3)
        #pragma unroll
        for (int r = 0; r < 8; ++r) {
            float c_ = cf[r + 8 * (lane >> 4)];
            #pragma unroll
            for (int ct = 0; ct < 4; ++ct) Oacc[ct][r] *= c_;
        }
        #pragma unroll
        for (int kj = 0; kj < 4; ++kj) {         // K = 128 keys in 32-steps
            v16bf ap = *(const v16bf*)&Ps[lr * PS_LD + kj * 32 + lh];
            #pragma unroll
            for (int ct = 0; ct < 4; ++ct) {
                v16bf bv2 = *(const v16bf*)&Vs[((w * 4 + ct) * 16 + lr) * VS_LD
                                               + kj * 32 + lh];
                Oacc[ct] = wmma_bf16(ap, bv2, Oacc[ct]);
            }
        }
    }
    __syncthreads();

    bf16* ob = oT + ((size_t)b * AREA + i0) * Cc;
    #pragma unroll
    for (int r = 0; r < 8; ++r) {
        int m = r + 8 * (lane >> 4);
        float inv = 1.f / s_sh[m];
        #pragma unroll
        for (int ct = 0; ct < 4; ++ct) {
            int n = (w * 4 + ct) * 16 + lr;
            ob[(size_t)m * Cc + n] = (bf16)(Oacc[ct][r] * inv);
        }
    }
}

// ---------------------------------------------------------------------------
// Kernel 5: output projection + bias + residual.
// resT[a][o] = oT[b][a][:] . wo[o][:] + bo[o];  out[b][o][a] = net + resT.
// D tile is staged through LDS so the final [c][a] fp32 stores are coalesced.
// ---------------------------------------------------------------------------
constexpr size_t PROJ_SMEM_BYTES = GEMM_DB_BYTES + (size_t)(128 * 132) * 4;

__global__ __launch_bounds__(256)
void proj_o_residual(const bf16* __restrict__ oT, const bf16* __restrict__ wo_bf,
                     const float* __restrict__ bo, const float* __restrict__ net,
                     float* __restrict__ out) {
    extern __shared__ char smem[];
    bf16* As0 = (bf16*)smem;
    bf16* Bs0 = As0 + 128 * 72;
    bf16* As1 = Bs0 + 128 * 72;
    bf16* Bs1 = As1 + 128 * 72;
    float* tileD = (float*)(Bs1 + 128 * 72);

    const int b  = blockIdx.z;
    const int m0 = blockIdx.x * 128;   // a
    const int n0 = blockIdx.y * 128;   // o

    v8f acc[8];
    gemm128_db(oT + (size_t)b * AREA * Cc, wo_bf, m0, n0, As0, Bs0, As1, Bs1, acc);

    const int tid  = threadIdx.x;
    const int lane = tid & 31;
    const int w    = tid >> 5;
    const int lr   = lane & 15;
    #pragma unroll
    for (int nt = 0; nt < 8; ++nt) {
        #pragma unroll
        for (int r = 0; r < 8; ++r) {
            int m = w * 16 + r + 8 * (lane >> 4);
            int n = nt * 16 + lr;
            tileD[m * 132 + n] = acc[nt][r] + bo[n0 + n];
        }
    }
    __syncthreads();

    const float* nb = net + (size_t)b * Cc * AREA;
    float*       ob = out + (size_t)b * Cc * AREA;
    for (int i = 0; i < 64; ++i) {               // 128*128 / 256
        int idx = tid + i * 256;
        int n = idx >> 7, m = idx & 127;
        size_t g = (size_t)(n0 + n) * AREA + (m0 + m);
        ob[g] = nb[g] + tileD[m * 132 + n];
    }
}

// ---------------------------------------------------------------------------
// host launcher
// ---------------------------------------------------------------------------
extern "C" void kernel_launch(void* const* d_in, const int* in_sizes, int n_in,
                              void* d_out, int out_size, void* d_ws, size_t ws_size,
                              hipStream_t stream) {
    const float* net    = (const float*)d_in[0];
    const float* gscale = (const float*)d_in[1];
    const float* gbias  = (const float*)d_in[2];
    const float* wq     = (const float*)d_in[3];
    const float* bq     = (const float*)d_in[4];
    const float* wk     = (const float*)d_in[5];
    const float* bk     = (const float*)d_in[6];
    const float* wv     = (const float*)d_in[7];
    const float* bv     = (const float*)d_in[8];
    const float* wo     = (const float*)d_in[9];
    const float* bo     = (const float*)d_in[10];
    float* out = (float*)d_out;

    // workspace carve-up (bf16 unless noted)
    const size_t S1 = (size_t)Bn * AREA * Cc;     // 8,388,608 elements
    bf16* xnT = (bf16*)d_ws;                      // [b][a][c]
    bf16* qT  = xnT + S1;                         // [b][a][c]
    bf16* kT  = qT + S1;                          // [b][a][c]
    bf16* vM  = kT + S1;                          // [b][c][a]
    bf16* oT  = vM + S1;                          // [b][a][c]
    bf16* wbf = oT + S1;                          // wq|wk|wv|wo, each [o][c]
    float* stats = (float*)(wbf + (size_t)4 * Cc * Cc);  // mean|rstd, 2*BG f32

    convert_weights<<<dim3(1024, 4), 256, 0, stream>>>(wq, wk, wv, wo, wbf);
    gn_stats<<<dim3(BG), 256, 0, stream>>>(net, stats);
    gn_apply_transpose<<<dim3(AREA / 64, Cc / 64, Bn), 256, 0, stream>>>(
        net, gscale, gbias, stats, xnT);

    (void)hipFuncSetAttribute((const void*)qkv_gemm,
                              hipFuncAttributeMaxDynamicSharedMemorySize,
                              (int)GEMM_DB_BYTES);
    qkv_gemm<<<dim3(AREA / 128, Cc / 128, Bn * 3), 256, GEMM_DB_BYTES, stream>>>(
        xnT, wbf, bq, bk, bv, qT, kT, vM);

    (void)hipFuncSetAttribute((const void*)flash_attn,
                              hipFuncAttributeMaxDynamicSharedMemorySize,
                              (int)ATTN_SMEM_BYTES);
    flash_attn<<<dim3(AREA / 16, Bn), 256, ATTN_SMEM_BYTES, stream>>>(
        qT, kT, vM, oT);

    (void)hipFuncSetAttribute((const void*)proj_o_residual,
                              hipFuncAttributeMaxDynamicSharedMemorySize,
                              (int)PROJ_SMEM_BYTES);
    proj_o_residual<<<dim3(AREA / 128, Cc / 128, Bn), 256, PROJ_SMEM_BYTES, stream>>>(
        oT, wbf + (size_t)3 * Cc * Cc, bo, net, out);
}
